// Sound_Classification_MLP_21904333209753
// MI455X (gfx1250) — 8-bit matrix operand order repaired and checked by reading, compiled, not run on hardware
//
#include <hip/hip_runtime.h>
#include <stdint.h>

typedef __attribute__((ext_vector_type(8))) int v8i;
typedef __attribute__((ext_vector_type(4))) int v4i32;

#define B_ROWS 32768
#define K1     1280
#define N1     500
#define N1P    512
#define K2     512
#define NOUT   10

#define M_BLK  256
#define N_BLK  128
#define NKS    (K1 / 64)

#define SX_OFF   ((size_t)0)
#define SW1_OFF  (SX_OFF  + (size_t)B_ROWS * K1)
#define SW2_OFF  (SW1_OFF + (size_t)N1P * K1)
#define A_OFF    (SW2_OFF + (size_t)16 * K2)
#define C_OFF    (A_OFF   + (size_t)N1P * 4)
#define H_OFF    (C_OFF   + (size_t)N1P * 4)

#if defined(__gfx1250__) && __has_builtin(__builtin_amdgcn_global_load_async_to_lds_b128)
#define HAS_ASYNC_LDS 1
typedef __attribute__((address_space(1))) v4i32 g_v4i32;
typedef __attribute__((address_space(3))) v4i32 l_v4i32;
#define ASYNC_CP_B128(gsrc, ldst)                                             \
    __builtin_amdgcn_global_load_async_to_lds_b128(                           \
        (g_v4i32*)(gsrc), (l_v4i32*)(ldst), 0, 0)
#if __has_builtin(__builtin_amdgcn_s_wait_asynccnt)
#define WAIT_ASYNC() __builtin_amdgcn_s_wait_asynccnt(0)
#else
#define WAIT_ASYNC() asm volatile("s_wait_asynccnt 0x0" ::: "memory")
#endif
#else
#define HAS_ASYNC_LDS 0
#endif

__device__ __forceinline__ int8_t signf8(float v) {
    return (int8_t)((v > 0.f) - (v < 0.f));
}

__global__ __launch_bounds__(256) void k_bin_x(const float* __restrict__ x,
                                               int8_t* __restrict__ sx, int n4) {
    int i = blockIdx.x * blockDim.x + threadIdx.x;
    if (i >= n4) return;
    float4 v = ((const float4*)x)[i];
    int32_t p;
    int8_t* pb = (int8_t*)&p;
    pb[0] = signf8(v.x); pb[1] = signf8(v.y);
    pb[2] = signf8(v.z); pb[3] = signf8(v.w);
    ((int32_t*)sx)[i] = p;
}

__global__ __launch_bounds__(256) void k_prep_w1(const float* __restrict__ W1,
                                                 int8_t* __restrict__ sw) {
    int d = blockIdx.x * blockDim.x + threadIdx.x;
    if (d >= N1P * (K1 / 4)) return;
    int row = d / (K1 / 4);
    int c0  = (d % (K1 / 4)) * 4;
    int32_t p = 0;
    if (row < N1) {
        int8_t* pb = (int8_t*)&p;
        #pragma unroll
        for (int j = 0; j < 4; ++j) pb[j] = signf8(W1[(size_t)row * K1 + c0 + j]);
    }
    ((int32_t*)sw)[d] = p;
}

__global__ __launch_bounds__(256) void k_prep_w2(const float* __restrict__ W2,
                                                 int8_t* __restrict__ sw) {
    int d = blockIdx.x * blockDim.x + threadIdx.x;
    if (d >= 16 * (K2 / 4)) return;
    int row = d / (K2 / 4);
    int c0  = (d % (K2 / 4)) * 4;
    int32_t p = 0;
    if (row < NOUT) {
        int8_t* pb = (int8_t*)&p;
        #pragma unroll
        for (int j = 0; j < 4; ++j) {
            int c = c0 + j;
            if (c < N1) pb[j] = signf8(W2[(size_t)row * N1 + c]);
        }
    }
    ((int32_t*)sw)[d] = p;
}

__global__ __launch_bounds__(256) void k_gemm1(const int8_t* __restrict__ sx,
                                               const int8_t* __restrict__ sw1,
                                               int32_t* __restrict__ hInt) {
    __shared__ int8_t ldsA[2][M_BLK * 64];
    __shared__ int8_t ldsB[2][N_BLK * 64];
    const int t       = threadIdx.x;
    const int rowBase = (int)(blockIdx.x >> 2) * M_BLK;
    const int colBase = (int)(blockIdx.x & 3) * N_BLK;
    const int lane = t & 31, w = t >> 5;
    const int wm = (w >> 1) * 64;
    const int wn = (w & 1) * 64;
    const int nl = lane & 15, hi = lane >> 4;

    const v8i vzero = {0, 0, 0, 0, 0, 0, 0, 0};
    v8i acc[4][4];
    #pragma unroll
    for (int mi = 0; mi < 4; ++mi)
        #pragma unroll
        for (int ni = 0; ni < 4; ++ni) acc[mi][ni] = vzero;

    auto stage = [&](int buf, int k0) {
#if HAS_ASYNC_LDS
        #pragma unroll
        for (int i = 0; i < 4; ++i) {
            const int e = t + i * 256, row = e >> 2, seg = (e & 3) * 16;
            ASYNC_CP_B128(sx + (size_t)(rowBase + row) * K1 + k0 + seg,
                          &ldsA[buf][row * 64 + seg]);
        }
        #pragma unroll
        for (int i = 0; i < 2; ++i) {
            const int e = t + i * 256, row = e >> 2, seg = (e & 3) * 16;
            ASYNC_CP_B128(sw1 + (size_t)(colBase + row) * K1 + k0 + seg,
                          &ldsB[buf][row * 64 + seg]);
        }
#else
        int4 ra[4], rb[2];
        #pragma unroll
        for (int i = 0; i < 4; ++i) {
            const int e = t + i * 256, row = e >> 2, seg = (e & 3) * 16;
            ra[i] = *(const int4*)(sx + (size_t)(rowBase + row) * K1 + k0 + seg);
        }
        #pragma unroll
        for (int i = 0; i < 2; ++i) {
            const int e = t + i * 256, row = e >> 2, seg = (e & 3) * 16;
            rb[i] = *(const int4*)(sw1 + (size_t)(colBase + row) * K1 + k0 + seg);
        }
        #pragma unroll
        for (int i = 0; i < 4; ++i) {
            const int e = t + i * 256, row = e >> 2, seg = (e & 3) * 16;
            *(int4*)&ldsA[buf][row * 64 + seg] = ra[i];
        }
        #pragma unroll
        for (int i = 0; i < 2; ++i) {
            const int e = t + i * 256, row = e >> 2, seg = (e & 3) * 16;
            *(int4*)&ldsB[buf][row * 64 + seg] = rb[i];
        }
#endif
    };

    stage(0, 0);
    for (int ks = 0; ks < NKS; ++ks) {
        const int cur = ks & 1;
#if HAS_ASYNC_LDS
        WAIT_ASYNC();
#endif
        __syncthreads();
        if (ks + 1 < NKS) stage(cur ^ 1, (ks + 1) * 64);

        v8i af[4];
        #pragma unroll
        for (int mi = 0; mi < 4; ++mi)
            #pragma unroll
            for (int v = 0; v < 8; ++v) {
                const int kk = ((v >> 1) & 1) * 16 + ((v >> 2) & 1) * 32 + hi * 8 + (v & 1) * 4;
                af[mi][v] = *(const int32_t*)&ldsA[cur][(wm + mi * 16 + nl) * 64 + kk];
            }
        #pragma unroll
        for (int ni = 0; ni < 4; ++ni) {
            v8i bf;
            #pragma unroll
            for (int v = 0; v < 8; ++v) {
                const int kk = ((v >> 1) & 1) * 16 + ((v >> 2) & 1) * 32 + hi * 8 + (v & 1) * 4;
                bf[v] = *(const int32_t*)&ldsB[cur][(wn + ni * 16 + nl) * 64 + kk];
            }
            #pragma unroll
            for (int mi = 0; mi < 4; ++mi)
                acc[mi][ni] = __builtin_amdgcn_wmma_i32_16x16x64_iu8(
                    true, af[mi], true, bf, acc[mi][ni], false, false);
        }
        __syncthreads();
    }

    #pragma unroll
    for (int mi = 0; mi < 4; ++mi)
        #pragma unroll
        for (int ni = 0; ni < 4; ++ni)
            #pragma unroll
            for (int v = 0; v < 8; ++v) {
                const int m = rowBase + wm + mi * 16 + v + hi * 8;
                const int n = colBase + wn + ni * 16 + nl;
                hInt[(size_t)m * N1P + n] = acc[mi][ni][v];
            }
}

__global__ __launch_bounds__(256) void k_stats(const int32_t* __restrict__ hInt,
                                               const float* __restrict__ gamma,
                                               const float* __restrict__ beta,
                                               float* __restrict__ aArr,
                                               float* __restrict__ cArr) {
    __shared__ double sh_s[256];
    __shared__ double sh_q[256];
    const int f = blockIdx.x;
    const int t = threadIdx.x;
    double s = 0.0, q = 0.0;
    for (int r = t; r < B_ROWS; r += 256) {
        const double v = (double)hInt[(size_t)r * N1P + f];
        s += v; q += v * v;
    }
    sh_s[t] = s; sh_q[t] = q;
    __syncthreads();
    for (int off = 128; off > 0; off >>= 1) {
        if (t < off) { sh_s[t] += sh_s[t + off]; sh_q[t] += sh_q[t + off]; }
        __syncthreads();
    }
    if (t == 0) {
        const double mean = sh_s[0] / (double)B_ROWS;
        const double var  = sh_q[0] / (double)B_ROWS - mean * mean;
        float g = 0.f, be = 0.f;
        if (f < N1) { g = gamma[f]; be = beta[f]; }
        const float a = g * rsqrtf((float)var + 1e-5f);
        aArr[f] = a;
        cArr[f] = be - (float)mean * a;
    }
}

__global__ __launch_bounds__(256) void k_layer2(const int32_t* __restrict__ hInt,
                                                const float* __restrict__ aArr,
                                                const float* __restrict__ cArr,
                                                const int8_t* __restrict__ sw2,
                                                const float* __restrict__ b2,
                                                float* __restrict__ out) {
    __shared__ int8_t ldsZ[128 * K2];
    const int t = threadIdx.x;
    const int rowBase = (int)blockIdx.x * 128;

    for (int i = t; i < 128 * (K2 / 4); i += 256) {
        const int row = i >> 7;
        const int c0  = (i & 127) * 4;
        const int32_t* hr = &hInt[(size_t)(rowBase + row) * N1P + c0];
        int32_t p = 0;
        int8_t* pb = (int8_t*)&p;
        #pragma unroll
        for (int j = 0; j < 4; ++j) {
            const float hn = (float)hr[j] * aArr[c0 + j] + cArr[c0 + j];
            pb[j] = hn > 0.f ? (int8_t)1 : (int8_t)0;
        }
        ((int32_t*)ldsZ)[i] = p;
    }
    __syncthreads();

    const int lane = t & 31, w = t >> 5;
    const int nl = lane & 15, hi = lane >> 4;
    v8i acc = {0, 0, 0, 0, 0, 0, 0, 0};
    for (int ks = 0; ks < K2 / 64; ++ks) {
        v8i af, bf;
        #pragma unroll
        for (int v = 0; v < 8; ++v) {
            const int ka = ks * 64 + ((v >> 1) & 1) * 16 + ((v >> 2) & 1) * 32 + hi * 8 + (v & 1) * 4;
            af[v] = *(const int32_t*)&ldsZ[(w * 16 + nl) * K2 + ka];
            const int kb = ks * 64 + ((v >> 1) & 1) * 16 + ((v >> 2) & 1) * 32 + hi * 8 + (v & 1) * 4;
            bf[v] = *(const int32_t*)&sw2[(size_t)nl * K2 + kb];
        }
        acc = __builtin_amdgcn_wmma_i32_16x16x64_iu8(
              false, af,   true, bf, acc, false, false);
    }
    if (nl < NOUT) {
        const float bias = b2[nl];
        #pragma unroll
        for (int v = 0; v < 8; ++v) {
            const int m = rowBase + w * 16 + v + hi * 8;
            out[(size_t)m * NOUT + nl] = (float)acc[v] + bias;
        }
    }
}

extern "C" void kernel_launch(void* const* d_in, const int* in_sizes, int n_in,
                              void* d_out, int out_size, void* d_ws, size_t ws_size,
                              hipStream_t stream) {
    const float* x     = (const float*)d_in[0];
    const float* W1    = (const float*)d_in[1];
    const float* gamma = (const float*)d_in[3];
    const float* beta  = (const float*)d_in[4];
    const float* W2    = (const float*)d_in[5];
    const float* b2    = (const float*)d_in[6];
    float* out = (float*)d_out;

    char* ws = (char*)d_ws;
    int8_t*  sx   = (int8_t*)(ws + SX_OFF);
    int8_t*  sw1  = (int8_t*)(ws + SW1_OFF);
    int8_t*  sw2  = (int8_t*)(ws + SW2_OFF);
    float*   aArr = (float*)(ws + A_OFF);
    float*   cArr = (float*)(ws + C_OFF);
    int32_t* hInt = (int32_t*)(ws + H_OFF);

    const int n4x = B_ROWS * K1 / 4;
    k_bin_x<<<(n4x + 255) / 256, 256, 0, stream>>>(x, sx, n4x);
    k_prep_w1<<<(N1P * (K1 / 4) + 255) / 256, 256, 0, stream>>>(W1, sw1);
    k_prep_w2<<<(16 * (K2 / 4) + 255) / 256, 256, 0, stream>>>(W2, sw2);

    k_gemm1<<<(B_ROWS / M_BLK) * (N1P / N_BLK), 256, 0, stream>>>(sx, sw1, hInt);

    k_stats<<<N1P, 256, 0, stream>>>(hInt, gamma, beta, aArr, cArr);

    k_layer2<<<B_ROWS / 128, 256, 0, stream>>>(hInt, aArr, cArr, sw2, b2, out);
}
